// RNNNew_81338090651731
// MI455X (gfx1250) — compile-verified, matrix operand order checked by reading, not run on hardware
//
#include <hip/hip_runtime.h>
#include <hip/hip_bf16.h>

// ---------------------------------------------------------------------------
// 2-layer tanh RNN for MI455X (gfx1250, wave32, WMMA bf16 / f32-accum).
//   Layer l:  xW = X @ W_ih + (b_ih + b_hh)        (big parallel GEMM, WMMA)
//             h_t = tanh(xW_t + h_{t-1} @ W_hh)    (persistent single-WGP scan)
// A-operands pre-converted to bf16 so GEMM inner loop is pure b128-load+WMMA.
// ---------------------------------------------------------------------------

typedef __attribute__((ext_vector_type(16))) __bf16 v16bf;
typedef __attribute__((ext_vector_type(8)))  __bf16 v8bf;
typedef __attribute__((ext_vector_type(8)))  float  v8f;

#define TT 1024
#define BB 64
#define HH 512
#define LDS_STRIDE 520   // bf16 elements per LDS row (16B-aligned, bank-skewed)

// ---- fragment helpers (layouts per CDNA5 ISA 7.12.2, wave32) ---------------

__device__ __forceinline__ v16bf cat8(v8bf lo, v8bf hi) {
  return __builtin_shufflevector(lo, hi, 0,1,2,3,4,5,6,7,8,9,10,11,12,13,14,15);
}

__device__ __forceinline__ v8bf cvt8(const float* p) {
  v8bf r;
#pragma unroll
  for (int i = 0; i < 8; ++i) r[i] = (__bf16)p[i];
  return r;
}

// 16x32 bf16 fragment from a bf16 row-major matrix (A from global/LDS, or
// B via pre-transposed W^T). Two contiguous 16B chunks -> b128 loads.
// lane 0-15 : major row r,  K = [k0..k0+7]  and [k0+16..k0+23]
// lane 16-31: major row r,  K = [k0+8..k0+15] and [k0+24..k0+31]
__device__ __forceinline__ v16bf load_frag_bf16(const __bf16* Bm, int ldb,
                                                int maj0, int k0, int lane) {
  int r = lane & 15, hf = lane >> 4;
  const __bf16* p = Bm + (size_t)(maj0 + r) * ldb + (k0 + hf * 8);
  v8bf lo = *(const v8bf*)p;
  v8bf hi = *(const v8bf*)(p + 16);
  return cat8(lo, hi);
}

__device__ __forceinline__ v8f wmma_bf16(v16bf a, v16bf b, v8f c) {
  return __builtin_amdgcn_wmma_f32_16x16x32_bf16(
      /*neg_a=*/false, a, /*neg_b=*/false, b,
      /*c_mod=*/(short)0, c, /*reuse_a=*/false, /*reuse_b=*/false);
}

// tanh via v_exp_f32 + v_rcp_f32 (TRANS ops co-execute with the XDL pipe).
// tanh(x) = 1 - 2/(e^{2x}+1);  e^{2x} = 2^{x * 2*log2(e)}
__device__ __forceinline__ float fast_tanh(float x) {
  float z = __builtin_amdgcn_exp2f(x * 2.8853900817779268f);
  return 1.0f - 2.0f * __builtin_amdgcn_rcpf(z + 1.0f);
}

// ---- kernel 0a: weight transpose + bf16 convert, bias fuse -----------------

__global__ void rnn_prep(const float* __restrict__ Wi0, const float* __restrict__ Wh0,
                         const float* __restrict__ Wi1, const float* __restrict__ Wh1,
                         const float* __restrict__ bi0, const float* __restrict__ bh0,
                         const float* __restrict__ bi1, const float* __restrict__ bh1,
                         __bf16* __restrict__ WiT0, __bf16* __restrict__ WhT0,
                         __bf16* __restrict__ WiT1, __bf16* __restrict__ WhT1,
                         float* __restrict__ bias0, float* __restrict__ bias1) {
  int idx = blockIdx.x * blockDim.x + threadIdx.x;   // 512*512 threads
  int k = idx >> 9;
  int n = idx & 511;
  size_t tr = (size_t)n * HH + k;
  WiT0[tr] = (__bf16)Wi0[idx];
  WhT0[tr] = (__bf16)Wh0[idx];
  WiT1[tr] = (__bf16)Wi1[idx];
  WhT1[tr] = (__bf16)Wh1[idx];
  if (idx < HH) {
    bias0[idx] = bi0[idx] + bh0[idx];
    bias1[idx] = bi1[idx] + bh1[idx];
  }
}

// ---- kernel 0b: bulk f32 -> bf16 conversion (x activations) ----------------

__global__ __launch_bounds__(256)
void f32_to_bf16(const float* __restrict__ src, __bf16* __restrict__ dst) {
  size_t base = ((size_t)blockIdx.x * blockDim.x + threadIdx.x) * 8;
  *(v8bf*)(dst + base) = cvt8(src + base);
}

// ---- kernel 1: big GEMM  C[M,512] = Abf[M,512] @ W + bias ------------------
// One wave computes a 32(M) x 64(N) tile: 2x4 WMMA accumulators, K-loop of 16.
// Inner loop: 12 b128 loads -> 8 v_wmma (A bf16 streamed, W^T L2-resident).

__global__ __launch_bounds__(256)
void rnn_gemm_bias(const __bf16* __restrict__ Abf, const __bf16* __restrict__ WT,
                   const float* __restrict__ bias, float* __restrict__ C, int M) {
  const int K = HH, N = HH;
  int lane = threadIdx.x & 31;
  int wave = (blockIdx.x * blockDim.x + threadIdx.x) >> 5;
  const int MT = M >> 5;                 // number of 32-row tiles
  int tm = wave % MT;
  int tn = wave / MT;
  if (tn >= N / 64) return;
  int row0 = tm * 32, col0 = tn * 64;

  v8f acc[2][4];
#pragma unroll
  for (int i = 0; i < 2; ++i)
#pragma unroll
    for (int j = 0; j < 4; ++j) acc[i][j] = (v8f)0.0f;

  for (int k0 = 0; k0 < K; k0 += 32) {
    v16bf a0 = load_frag_bf16(Abf, K, row0, k0, lane);
    v16bf a1 = load_frag_bf16(Abf, K, row0 + 16, k0, lane);
#pragma unroll
    for (int j = 0; j < 4; ++j) {
      v16bf b = load_frag_bf16(WT, K, col0 + j * 16, k0, lane);
      acc[0][j] = wmma_bf16(a0, b, acc[0][j]);
      acc[1][j] = wmma_bf16(a1, b, acc[1][j]);
    }
  }

  int c = lane & 15, hf = lane >> 4;
#pragma unroll
  for (int i2 = 0; i2 < 2; ++i2)
#pragma unroll
    for (int j = 0; j < 4; ++j) {
      int col = col0 + j * 16 + c;
      float bv = bias[col];
#pragma unroll
      for (int i = 0; i < 8; ++i) {
        int r = row0 + i2 * 16 + hf * 8 + i;
        C[(size_t)r * N + col] = acc[i2][j][i] + bv;
      }
    }
}

// ---- kernel 2: sequential recurrence, one persistent workgroup -------------
// 1024 threads = 32 waves on one WGP. Wave w owns output columns [16w,16w+16).
// h kept in double-buffered LDS (bf16); W_hh^T streams from L2 each step.
// P holds xW (f32) and is read each step. Outputs:
//   WRITE_F32 : y written back into P (layer-1 -> d_out H1)
//   WRITE_BF16: y written to Hbf (layer-0 -> bf16 A for layer-1 GEMM)
// Final hidden state (t == T-1) always written to hT (f32).

template <bool WRITE_F32, bool WRITE_BF16>
__global__ __launch_bounds__(1024)
void rnn_scan(float* __restrict__ P, const __bf16* __restrict__ WhT,
              const float* __restrict__ h0, float* __restrict__ hT,
              __bf16* __restrict__ Hbf, int T) {
  extern __shared__ __bf16 hbuf[];       // 2 * BB * LDS_STRIDE bf16
  int tid = threadIdx.x;
  int lane = tid & 31, wave = tid >> 5;
  int n0 = wave * 16;
  int c = lane & 15, hf = lane >> 4;

  // load initial hidden state (f32 -> bf16) into buffer 0
  for (int i = tid; i < BB * HH; i += 1024) {
    int r = i >> 9, col = i & 511;
    hbuf[r * LDS_STRIDE + col] = (__bf16)h0[i];
  }
  __syncthreads();

  for (int t = 0; t < T; ++t) {
    const __bf16* cur = hbuf + (size_t)(t & 1) * BB * LDS_STRIDE;
    __bf16* nxt = hbuf + (size_t)((t + 1) & 1) * BB * LDS_STRIDE;

    v8f acc[4];
#pragma unroll
    for (int m = 0; m < 4; ++m) acc[m] = (v8f)0.0f;

    for (int k0 = 0; k0 < HH; k0 += 32) {
      v16bf b = load_frag_bf16(WhT, HH, n0, k0, lane);   // L2-resident weights
#pragma unroll
      for (int m = 0; m < 4; ++m) {
        v16bf a = load_frag_bf16(cur, LDS_STRIDE, m * 16, k0, lane);
        acc[m] = wmma_bf16(a, b, acc[m]);
      }
    }

    float* Pt = P + (size_t)t * BB * HH;
    __bf16* Ht = WRITE_BF16 ? (Hbf + (size_t)t * BB * HH) : nullptr;
    int col = n0 + c;
#pragma unroll
    for (int m = 0; m < 4; ++m) {
#pragma unroll
      for (int i = 0; i < 8; ++i) {
        int r = m * 16 + hf * 8 + i;
        size_t idx = (size_t)r * HH + col;
        float y = fast_tanh(Pt[idx] + acc[m][i]);
        __bf16 yb = (__bf16)y;
        if (WRITE_F32) Pt[idx] = y;        // layer output H[t] (f32)
        if (WRITE_BF16) Ht[idx] = yb;      // bf16 copy for next layer's GEMM
        nxt[r * LDS_STRIDE + col] = yb;
        if (t == T - 1) hT[idx] = y;       // final hidden state
      }
    }
    __syncthreads();
  }
}

// ---------------------------------------------------------------------------

extern "C" void kernel_launch(void* const* d_in, const int* in_sizes, int n_in,
                              void* d_out, int out_size, void* d_ws, size_t ws_size,
                              hipStream_t stream) {
  (void)in_sizes; (void)n_in; (void)out_size; (void)ws_size;
  const float* x   = (const float*)d_in[0];
  const float* h_t = (const float*)d_in[1];
  const float* Wi0 = (const float*)d_in[2];
  const float* Wh0 = (const float*)d_in[3];
  const float* bi0 = (const float*)d_in[4];
  const float* bh0 = (const float*)d_in[5];
  const float* Wi1 = (const float*)d_in[6];
  const float* Wh1 = (const float*)d_in[7];
  const float* bi1 = (const float*)d_in[8];
  const float* bh1 = (const float*)d_in[9];
  float* out = (float*)d_out;

  const int M = TT * BB;                               // 65536 rows
  char* ws = (char*)d_ws;
  float*  P0   = (float*)ws;                           // [M,512] f32  (xW0)
  __bf16* Xbf  = (__bf16*)(ws + (size_t)M * HH * 4);   // [M,512] bf16 (x, then H0)
  __bf16* WiT0 = Xbf + (size_t)M * HH;
  __bf16* WhT0 = WiT0 + (size_t)HH * HH;
  __bf16* WiT1 = WhT0 + (size_t)HH * HH;
  __bf16* WhT1 = WiT1 + (size_t)HH * HH;
  float*  bias0 = (float*)(WhT1 + (size_t)HH * HH);
  float*  bias1 = bias0 + HH;
  __bf16* H0bf = Xbf;    // aliases Xbf: x no longer needed after layer-0 GEMM

  // 0) weights -> transposed bf16, fused biases; x -> bf16
  rnn_prep<<<(HH * HH) / 256, 256, 0, stream>>>(
      Wi0, Wh0, Wi1, Wh1, bi0, bh0, bi1, bh1,
      WiT0, WhT0, WiT1, WhT1, bias0, bias1);
  f32_to_bf16<<<(int)((size_t)M * HH / 8 / 256), 256, 0, stream>>>(x, Xbf);

  const int gemmBlocks = (M / 32) * (HH / 64) / 8;     // 8 waves per 256-thr block
  const size_t shBytes = 2 * BB * LDS_STRIDE * sizeof(__bf16);
  float* hT0 = out + (size_t)TT * BB * HH;
  float* hT1 = hT0 + (size_t)BB * HH;

  // layer 0: xW0 = Xbf @ Wi0 + b -> P0 ; scan -> H0 (bf16, overwrites Xbf slot)
  rnn_gemm_bias<<<gemmBlocks, 256, 0, stream>>>(Xbf, WiT0, bias0, P0, M);
  rnn_scan<false, true><<<1, 1024, shBytes, stream>>>(P0, WhT0, h_t, hT0, H0bf, TT);

  // layer 1: xW1 = H0bf @ Wi1 + b -> d_out ; scan in place -> H1 (f32)
  rnn_gemm_bias<<<gemmBlocks, 256, 0, stream>>>(H0bf, WiT1, bias1, out, M);
  rnn_scan<true, false><<<1, 1024, shBytes, stream>>>(out, WhT1,
                                                      h_t + (size_t)BB * HH,
                                                      hT1, nullptr, TT);
}
